// GKTTransformer_83021717831968
// MI455X (gfx1250) — matrix operand order repaired and checked by reading, not compiled, not run on hardware
//
#include <hip/hip_runtime.h>
#include <math.h>
#include <stdint.h>

#define CCH   256
#define HWSZ  4096
#define NVIEW 54
#define EPSL  1e-5f

typedef __attribute__((ext_vector_type(16))) __bf16 v16bf;
typedef __attribute__((ext_vector_type(8)))  __bf16 v8bf;
typedef __attribute__((ext_vector_type(8)))  float  v8f;
typedef __attribute__((ext_vector_type(4)))  unsigned int u32x4;
typedef __attribute__((ext_vector_type(8)))  int i32x8;
typedef __attribute__((ext_vector_type(4)))  int i32x4;

union V16 { v16bf v; v8bf h[2]; };

#define WMMA_BF16(A,B,Cacc) \
  __builtin_amdgcn_wmma_f32_16x16x32_bf16(false,(A),false,(B),(short)0,(Cacc),false,false)

// ---- fragment loaders (16B ds_load_b128 chunks, per documented VGPR layouts) ----
// A: 16x32 bf16. lanes 0-15: K {0..7,16..23}; lanes 16-31: K {8..15,24..31}. Row pitch 32 bf16.
static __device__ inline v16bf load_fragA(const __bf16* AL, int m, int half) {
  const v8bf* p = (const v8bf*)(AL + m * 32);
  V16 u; u.h[0] = p[half]; u.h[1] = p[half + 2]; return u.v;
}
// B: 32x16 bf16 staged transposed [n][k]. lanes 0-15: K {0..7,16..23}; lanes 16-31: K {8..15,24..31}.
static __device__ inline v16bf load_fragB(const __bf16* B, int n, int half, int pitch) {
  const v8bf* p = (const v8bf*)(B + n * pitch);
  V16 u; u.h[0] = p[half]; u.h[1] = p[half + 2]; return u.v;
}

// ---- TDM: DMA a 256x32 bf16 K-slab of the 256x256 weight tensor into LDS [256][32] ----
// Issued by wave 0 only (EXEC ignored by TDM; branch keeps it to one issue per WG).
// D# per CDNA5 ISA: row-major tile walk writes LDS contiguously,
// giving exactly the AL[m][kloc] layout the fragment loader expects.
static __device__ inline void tdm_load_A(const __bf16* Wb, __bf16* AL, int kk) {
  unsigned long long ga = (unsigned long long)(const void*)(Wb + kk * 32);
  unsigned int ldsOff = (unsigned int)(unsigned long long)(const void*)AL;
  u32x4 g0;
  g0[0] = 1u;                                        // count=1, user descriptor
  g0[1] = ldsOff;                                    // lds_addr (bytes)
  g0[2] = (unsigned int)ga;                          // global_addr[31:0]
  g0[3] = (unsigned int)((ga >> 32) & 0x01FFFFFFull) // global_addr[56:32]
          | (2u << 30);                              // type=2 ("image")
  i32x8 g1;
  g1[0] = 0x00010000;   // workgroup_mask=0, data_size=1 (2B), no barrier/iter/pad
  g1[1] = 256 << 16;    // tensor_dim0 = 256 (lo16 in bits 63:48)
  g1[2] = 256 << 16;    // tensor_dim0 hi=0 ; tensor_dim1 = 256 (lo16 in bits 95:80)
  g1[3] = 32 << 16;     // tensor_dim1 hi=0 ; tile_dim0 = 32
  g1[4] = 256;          // tile_dim1 = 256 ; tile_dim2 = 0
  g1[5] = 256;          // tensor_dim0_stride = 256 (elements)
  g1[6] = 0;            // stride hi ; tensor_dim1_stride lo = 0
  g1[7] = 0;
  i32x4 gz = {0, 0, 0, 0};
#if defined(__clang_major__) && __clang_major__ >= 23
  i32x8 z8 = {0, 0, 0, 0, 0, 0, 0, 0};
  __builtin_amdgcn_tensor_load_to_lds(g0, g1, gz, gz, z8, 0);
#else
  __builtin_amdgcn_tensor_load_to_lds(g0, g1, gz, gz, 0);
#endif
  __builtin_amdgcn_s_wait_tensorcnt(0);
}

// stage activation (f32 global [c][HW], pre-offset to hw0) K-slab kk into BL [hw][k] bf16
// one global_load_b128 per thread (32 consecutive hw per row), transposed ds_store_b16 x4
static __device__ inline void stage_Bf32(const float* X, __bf16* BL, int kk, int tid) {
  int c = tid >> 3, h4 = (tid & 7) * 4;
  const float4 v4 = *(const float4*)(X + (size_t)((kk << 5) + c) * HWSZ + h4);
  BL[(h4 + 0) * 32 + c] = (__bf16)v4.x;
  BL[(h4 + 1) * 32 + c] = (__bf16)v4.y;
  BL[(h4 + 2) * 32 + c] = (__bf16)v4.z;
  BL[(h4 + 3) * 32 + c] = (__bf16)v4.w;
}

// ---------------- weight convert ----------------
__global__ void k_f2bf(const float* __restrict__ s, __bf16* __restrict__ d, int n) {
  int i = blockIdx.x * 256 + threadIdx.x;
  if (i < n) d[i] = (__bf16)s[i];
}

// ---------------- layernorm over C per spatial position ----------------
__global__ void k_layernorm(const float* __restrict__ x, const float* __restrict__ g,
                            const float* __restrict__ b, float* __restrict__ y) {
  __shared__ float ps[8][32], pq[8][32], mS[32], rS[32];
  int tid = threadIdx.x, tx = tid & 31, ty = tid >> 5;
  int hw = blockIdx.x * 32 + tx;
  float vals[32]; float s = 0.f, q2 = 0.f;
#pragma unroll
  for (int i = 0; i < 32; ++i) {
    float v_ = x[(size_t)(ty * 32 + i) * HWSZ + hw];
    vals[i] = v_; s += v_; q2 += v_ * v_;
  }
  ps[ty][tx] = s; pq[ty][tx] = q2;
  __syncthreads();
  if (ty == 0) {
    float S = 0.f, Q = 0.f;
#pragma unroll
    for (int r = 0; r < 8; ++r) { S += ps[r][tx]; Q += pq[r][tx]; }
    float m = S * (1.f / CCH);
    float var = Q * (1.f / CCH) - m * m;
    mS[tx] = m; rS[tx] = rsqrtf(var + EPSL);
  }
  __syncthreads();
  float m = mS[tx], r = rS[tx];
#pragma unroll
  for (int i = 0; i < 32; ++i) {
    int c = ty * 32 + i;
    y[(size_t)c * HWSZ + hw] = (vals[i] - m) * r * g[c] + b[c];
  }
}

// ---------------- qp = Wq (bf16) @ norm_q : full M=256 x 32-hw tile per block ----------------
__global__ void k_gemm_qp(const __bf16* __restrict__ Wb, const float* __restrict__ X,
                          float* __restrict__ Y) {
  extern __shared__ char smem[];
  __bf16* AL = (__bf16*)smem;            // 16384
  __bf16* BL = (__bf16*)(smem + 16384);  // 2048
  int tid = threadIdx.x, lane = tid & 31, wave = tid >> 5;
  int half = lane >> 4, l15 = lane & 15;
  int hw0 = blockIdx.x * 32;
  const float* Xs = X + hw0;
  v8f acc[2][2] = {};
  for (int kk = 0; kk < 8; ++kk) {
    if (tid < 32) tdm_load_A(Wb, AL, kk);
    stage_Bf32(Xs, BL, kk, tid);
    __syncthreads();
    v16bf a0 = load_fragA(AL, wave * 32 + l15, half);
    v16bf a1 = load_fragA(AL, wave * 32 + 16 + l15, half);
    v16bf b0 = load_fragB(BL, l15, half, 32);
    v16bf b1 = load_fragB(BL, 16 + l15, half, 32);
    acc[0][0] = WMMA_BF16(a0, b0, acc[0][0]);
    acc[0][1] = WMMA_BF16(a0, b1, acc[0][1]);
    acc[1][0] = WMMA_BF16(a1, b0, acc[1][0]);
    acc[1][1] = WMMA_BF16(a1, b1, acc[1][1]);
    __syncthreads();
  }
#pragma unroll
  for (int mi = 0; mi < 2; ++mi)
#pragma unroll
    for (int si = 0; si < 2; ++si)
#pragma unroll
      for (int j = 0; j < 8; ++j) {
        int M = wave * 32 + mi * 16 + (half ? 8 : 0) + j;
        int h = si * 16 + l15;
        Y[(size_t)M * HWSZ + hw0 + h] = acc[mi][si][j];
      }
}

// ---------------- att: loop views, kp = Wk@k_n via WMMA, dot with qp, softmax over views ----------------
__global__ void k_att(const __bf16* __restrict__ Wkb, const float* __restrict__ Kin,
                      const float* __restrict__ qp, float* __restrict__ att) {
  extern __shared__ char smem[];
  __bf16* AL  = (__bf16*)smem;            // 16384
  __bf16* BL  = (__bf16*)(smem + 16384);  // 2048
  float*  qpL = (float*)(smem + 18432);   // 32768  [hw][c]
  float*  lg  = (float*)(smem + 51200);   // 6912   [n][hw]
  int tid = threadIdx.x, lane = tid & 31, wave = tid >> 5;
  int half = lane >> 4, l15 = lane & 15;
  int hw0 = blockIdx.x * 32;
  // stage qp tile transposed [hw][c] (coalesced along hw)
  {
    int hwc = tid & 31, cb = (tid >> 5) * 32;
#pragma unroll
    for (int i = 0; i < 32; ++i)
      qpL[hwc * 256 + cb + i] = qp[(size_t)(cb + i) * HWSZ + hw0 + hwc];
  }
  for (int idx = tid; idx < NVIEW * 32; idx += 256) lg[idx] = 0.f;
  __syncthreads();

  for (int n = 0; n < NVIEW; ++n) {
    const float* Xs = Kin + (size_t)n * CCH * HWSZ + hw0;
    v8f acc[2][2] = {};
    for (int kk = 0; kk < 8; ++kk) {
      if (tid < 32) tdm_load_A(Wkb, AL, kk);
      stage_Bf32(Xs, BL, kk, tid);
      __syncthreads();
      v16bf a0 = load_fragA(AL, wave * 32 + l15, half);
      v16bf a1 = load_fragA(AL, wave * 32 + 16 + l15, half);
      v16bf b0 = load_fragB(BL, l15, half, 32);
      v16bf b1 = load_fragB(BL, 16 + l15, half, 32);
      acc[0][0] = WMMA_BF16(a0, b0, acc[0][0]);
      acc[0][1] = WMMA_BF16(a0, b1, acc[0][1]);
      acc[1][0] = WMMA_BF16(a1, b0, acc[1][0]);
      acc[1][1] = WMMA_BF16(a1, b1, acc[1][1]);
      __syncthreads();
    }
    // logit partials: sum_c qp[c,hw]*kp[c,hw]
    float p0 = 0.f, p1 = 0.f;
#pragma unroll
    for (int mi = 0; mi < 2; ++mi)
#pragma unroll
      for (int j = 0; j < 8; ++j) {
        int M = wave * 32 + mi * 16 + (half ? 8 : 0) + j;
        p0 += acc[mi][0][j] * qpL[l15 * 256 + M];
        p1 += acc[mi][1][j] * qpL[(16 + l15) * 256 + M];
      }
    p0 += __shfl_xor(p0, 16);
    p1 += __shfl_xor(p1, 16);
    if (lane < 16) {
      atomicAdd(&lg[n * 32 + lane], p0);
      atomicAdd(&lg[n * 32 + 16 + lane], p1);
    }
  }
  __syncthreads();
  if (tid < 32) {
    float mx = -1e30f;
    for (int n = 0; n < NVIEW; ++n) mx = fmaxf(mx, lg[n * 32 + tid]);
    float s = 0.f;
    for (int n = 0; n < NVIEW; ++n) { float e = __expf(lg[n * 32 + tid] - mx); lg[n * 32 + tid] = e; s += e; }
    float inv = 1.f / s;
    for (int n = 0; n < NVIEW; ++n) att[(size_t)n * HWSZ + hw0 + tid] = lg[n * 32 + tid] * inv;
  }
}

// ---------------- fused: x = sum_n att*(Wv@v_n); t = LN2(x+norm_q); out = t + W2@relu(W1@t) ----------------
__global__ void k_out(const __bf16* __restrict__ Wvb, const __bf16* __restrict__ W1b,
                      const __bf16* __restrict__ W2b, const float* __restrict__ Vin,
                      const float* __restrict__ att, const float* __restrict__ nq,
                      const float* __restrict__ g2, const float* __restrict__ b2,
                      float* __restrict__ out) {
  extern __shared__ char smem[];
  __bf16* AL   = (__bf16*)smem;            // 16384, live always
  __bf16* BL   = (__bf16*)(smem + 16384);  // 2048,  phase 1 (view loop)
  float*  attL = (float*)(smem + 18432);   // 6912,  phase 1
  __bf16* tbfL = (__bf16*)(smem + 16384);  // 16384, phase 2 (aliases BL+attL): t bf16, then h bf16
  float*  tL   = (float*)(smem + 32768);   // 32768  t f32  -> total 65536
  int tid = threadIdx.x, lane = tid & 31, wave = tid >> 5;
  int half = lane >> 4, l15 = lane & 15;
  int hw0 = blockIdx.x * 32;

  for (int idx = tid; idx < NVIEW * 32; idx += 256)
    attL[idx] = att[(size_t)(idx >> 5) * HWSZ + hw0 + (idx & 31)];
  __syncthreads();

  v8f xacc[2][2] = {};
  for (int n = 0; n < NVIEW; ++n) {
    const float* Xs = Vin + (size_t)n * CCH * HWSZ + hw0;
    v8f acc[2][2] = {};
    for (int kk = 0; kk < 8; ++kk) {
      if (tid < 32) tdm_load_A(Wvb, AL, kk);
      stage_Bf32(Xs, BL, kk, tid);
      __syncthreads();
      v16bf a0 = load_fragA(AL, wave * 32 + l15, half);
      v16bf a1 = load_fragA(AL, wave * 32 + 16 + l15, half);
      v16bf b0 = load_fragB(BL, l15, half, 32);
      v16bf b1 = load_fragB(BL, 16 + l15, half, 32);
      acc[0][0] = WMMA_BF16(a0, b0, acc[0][0]);
      acc[0][1] = WMMA_BF16(a0, b1, acc[0][1]);
      acc[1][0] = WMMA_BF16(a1, b0, acc[1][0]);
      acc[1][1] = WMMA_BF16(a1, b1, acc[1][1]);
      __syncthreads();
    }
    float w0 = attL[n * 32 + l15];
    float w1 = attL[n * 32 + 16 + l15];
#pragma unroll
    for (int mi = 0; mi < 2; ++mi) {
      xacc[mi][0] += acc[mi][0] * w0;
      xacc[mi][1] += acc[mi][1] * w1;
    }
  }
  // residual add, park in LDS [hw][c]
#pragma unroll
  for (int mi = 0; mi < 2; ++mi)
#pragma unroll
    for (int si = 0; si < 2; ++si)
#pragma unroll
      for (int j = 0; j < 8; ++j) {
        int M = wave * 32 + mi * 16 + (half ? 8 : 0) + j;
        int h = si * 16 + l15;
        tL[h * 256 + M] = xacc[mi][si][j] + nq[(size_t)M * HWSZ + hw0 + h];
      }
  __syncthreads();
  // layernorm2 per column (wave per 4 columns), write f32 + bf16
#pragma unroll
  for (int co = 0; co < 4; ++co) {
    int col = wave * 4 + co;
    int base = col * 256 + lane * 8;
    float s = 0.f, q2 = 0.f;
#pragma unroll
    for (int e = 0; e < 8; ++e) { float v_ = tL[base + e]; s += v_; q2 += v_ * v_; }
#pragma unroll
    for (int m_ = 16; m_ >= 1; m_ >>= 1) { s += __shfl_xor(s, m_); q2 += __shfl_xor(q2, m_); }
    float mn = s * (1.f / CCH);
    float var = q2 * (1.f / CCH) - mn * mn;
    float rs = rsqrtf(var + EPSL);
#pragma unroll
    for (int e = 0; e < 8; ++e) {
      int c = lane * 8 + e;
      float v_ = (tL[base + e] - mn) * rs * g2[c] + b2[c];
      tL[base + e] = v_;
      tbfL[col * 256 + c] = (__bf16)v_;
    }
  }
  __syncthreads();
  // FFN1: h = relu(W1 @ t)
  v8f hacc[2][2] = {};
  for (int kk = 0; kk < 8; ++kk) {
    if (tid < 32) tdm_load_A(W1b, AL, kk);
    __syncthreads();
    v16bf a0 = load_fragA(AL, wave * 32 + l15, half);
    v16bf a1 = load_fragA(AL, wave * 32 + 16 + l15, half);
    v16bf b0 = load_fragB(tbfL + kk * 32, l15, half, 256);
    v16bf b1 = load_fragB(tbfL + kk * 32, 16 + l15, half, 256);
    hacc[0][0] = WMMA_BF16(a0, b0, hacc[0][0]);
    hacc[0][1] = WMMA_BF16(a0, b1, hacc[0][1]);
    hacc[1][0] = WMMA_BF16(a1, b0, hacc[1][0]);
    hacc[1][1] = WMMA_BF16(a1, b1, hacc[1][1]);
    __syncthreads();
  }
#pragma unroll
  for (int mi = 0; mi < 2; ++mi)
#pragma unroll
    for (int si = 0; si < 2; ++si)
#pragma unroll
      for (int j = 0; j < 8; ++j) {
        int M = wave * 32 + mi * 16 + (half ? 8 : 0) + j;
        int h = si * 16 + l15;
        tbfL[h * 256 + M] = (__bf16)fmaxf(hacc[mi][si][j], 0.f);  // h overwrites t-bf16
      }
  __syncthreads();
  // FFN2: t2 = W2 @ h ; out = t + t2
  v8f t2[2][2] = {};
  for (int kk = 0; kk < 8; ++kk) {
    if (tid < 32) tdm_load_A(W2b, AL, kk);
    __syncthreads();
    v16bf a0 = load_fragA(AL, wave * 32 + l15, half);
    v16bf a1 = load_fragA(AL, wave * 32 + 16 + l15, half);
    v16bf b0 = load_fragB(tbfL + kk * 32, l15, half, 256);
    v16bf b1 = load_fragB(tbfL + kk * 32, 16 + l15, half, 256);
    t2[0][0] = WMMA_BF16(a0, b0, t2[0][0]);
    t2[0][1] = WMMA_BF16(a0, b1, t2[0][1]);
    t2[1][0] = WMMA_BF16(a1, b0, t2[1][0]);
    t2[1][1] = WMMA_BF16(a1, b1, t2[1][1]);
    __syncthreads();
  }
#pragma unroll
  for (int mi = 0; mi < 2; ++mi)
#pragma unroll
    for (int si = 0; si < 2; ++si)
#pragma unroll
      for (int j = 0; j < 8; ++j) {
        int M = wave * 32 + mi * 16 + (half ? 8 : 0) + j;
        int h = si * 16 + l15;
        out[(size_t)M * HWSZ + hw0 + h] = tL[h * 256 + M] + t2[mi][si][j];
      }
}

extern "C" void kernel_launch(void* const* d_in, const int* in_sizes, int n_in,
                              void* d_out, int out_size, void* d_ws, size_t ws_size,
                              hipStream_t stream) {
  (void)in_sizes; (void)n_in; (void)out_size; (void)ws_size;
  const float* q  = (const float*)d_in[0];
  const float* k  = (const float*)d_in[1];
  const float* v  = (const float*)d_in[2];
  const float* Wq = (const float*)d_in[3];
  const float* Wk = (const float*)d_in[4];
  const float* Wv = (const float*)d_in[5];
  const float* W1 = (const float*)d_in[6];
  const float* W2 = (const float*)d_in[7];
  const float* g1 = (const float*)d_in[8];
  const float* b1 = (const float*)d_in[9];
  const float* g2 = (const float*)d_in[10];
  const float* b2 = (const float*)d_in[11];

  char* ws = (char*)d_ws;
  __bf16* Wqb = (__bf16*)(ws + 0);
  __bf16* Wkb = (__bf16*)(ws + 131072);
  __bf16* Wvb = (__bf16*)(ws + 262144);
  __bf16* W1b = (__bf16*)(ws + 393216);
  __bf16* W2b = (__bf16*)(ws + 524288);
  float* normq = (float*)(ws + 655360);   // 4 MB
  float* qp    = (float*)(ws + 4849664);  // 4 MB
  float* attw  = (float*)(ws + 9043968);  // 864 KB

  const int WN = CCH * CCH;
  k_f2bf<<<WN / 256, 256, 0, stream>>>(Wq, Wqb, WN);
  k_f2bf<<<WN / 256, 256, 0, stream>>>(Wk, Wkb, WN);
  k_f2bf<<<WN / 256, 256, 0, stream>>>(Wv, Wvb, WN);
  k_f2bf<<<WN / 256, 256, 0, stream>>>(W1, W1b, WN);
  k_f2bf<<<WN / 256, 256, 0, stream>>>(W2, W2b, WN);

  k_layernorm<<<HWSZ / 32, 256, 0, stream>>>(q, g1, b1, normq);
  k_gemm_qp<<<HWSZ / 32, 256, 18432, stream>>>(Wqb, normq, qp);
  k_att<<<HWSZ / 32, 256, 58112, stream>>>(Wkb, k, qp, attw);
  k_out<<<HWSZ / 32, 256, 65536, stream>>>(Wvb, W1b, W2b, v, attw, normq, g2, b2,
                                           (float*)d_out);
}
